// Seq2Seq_55430847922540
// MI455X (gfx1250) — matrix operand order repaired and checked by reading, not compiled, not run on hardware
//
#include <hip/hip_runtime.h>
#include <cmath>

typedef __attribute__((ext_vector_type(16))) __bf16 v16bf;
typedef __attribute__((ext_vector_type(8)))  __bf16 v8bf;
typedef __attribute__((ext_vector_type(8)))  float  v8f;

constexpr int  Bsz  = 1024;
constexpr int  Tlen = 128;
constexpr int  Hd   = 512;
constexpr int  Lnum = 4;
constexpr int  PRED = 10;
constexpr int  PITCH = 520;   // halfs per LDS row; keeps 16B alignment (1040B rows)
constexpr int  EROWS = 32;    // batch rows per WG (encoder)
constexpr int  DROWS = 16;    // batch rows per WG (decoder)

// ---------------- fragment helpers ----------------

// A fragment (16xK tile, bf16) from an LDS buffer with row pitch PITCH.
// Lane m (0-15) holds K = [0..15]; lane m+16 holds K = [16..31] (matches load_b).
__device__ inline v16bf load_a(const __bf16* buf, int mbase, int kb, int lane) {
    int m   = mbase + (lane & 15);
    int sel = lane >> 4;
    const __bf16* p = buf + m * PITCH + kb + sel * 16;
    v8bf lo = *(const v8bf*)p;
    v8bf hi = *(const v8bf*)(p + 8);
    v16bf r;
#pragma unroll
    for (int i = 0; i < 8; ++i) { r[i] = lo[i]; r[i + 8] = hi[i]; }
    return r;
}

// B fragment (Kx16 tile) for D = A * W^T : B[k][n] = W[n][k], W row-major [512][512].
// Lane n (0-15): K=0..15 ; lane n+16: K=16..31 -> one contiguous 32B read per lane.
__device__ inline v16bf load_b(const __bf16* W, int ncol, int kb, int lane) {
    int n   = ncol + (lane & 15);
    int sel = lane >> 4;
    return *(const v16bf*)(W + (size_t)n * Hd + kb + sel * 16);
}

__device__ inline v8f wmma_bf16(v16bf a, v16bf b, v8f c) {
    return __builtin_amdgcn_wmma_f32_16x16x32_bf16(
        /*neg_a=*/false, a, /*neg_b=*/false, b,
        /*c_mod=*/(short)0, c, /*reuse_a=*/false, /*reuse_b=*/false);
}

// Warm L2 with one weight matrix (512KB, 128B lines -> 64-half chunks).
__device__ inline void prefetch_w(const __bf16* W, int tid) {
    if (W) {
        for (int i = tid; i < (Hd * Hd) / 64; i += 256)
            __builtin_prefetch(W + i * 64, 0, 1);
    }
}

// ---------------- weight conversion f32 -> bf16 ----------------
__global__ void cvt_bf16(const float* __restrict__ s, __bf16* __restrict__ d, int n) {
    int i = blockIdx.x * 256 + threadIdx.x;
    if (i < n) d[i] = (__bf16)s[i];
}

// ---------------- encoder layer (time scan per WG) ----------------
// MODE 0: input = x f32 [B,T,6], out bf16 seq.   MODE 1: bf16 -> bf16 (in-place OK).
// MODE 2: bf16 -> f32 d_out (enc_outputs).
template <int MODE>
__global__ __launch_bounds__(256) void enc_layer(
    const void* __restrict__ inSeq, const float* __restrict__ wih0,
    const __bf16* __restrict__ Wih, const __bf16* __restrict__ Whh,
    const float* __restrict__ bih, const float* __restrict__ bhh,
    __bf16* __restrict__ outBf, float* __restrict__ outF,
    float* __restrict__ hT)
{
    extern __shared__ __align__(16) char smem[];
    __bf16* hA    = (__bf16*)smem;                 // 32x520 bf16 (h ping)
    __bf16* hB    = hA + EROWS * PITCH;            // h pong
    __bf16* xb0   = hB + EROWS * PITCH;            // x_t tile ping (bf16, MODE1/2)
    __bf16* xb1   = xb0 + EROWS * PITCH;           // x_t tile pong
    float*  biasS = (float*)(xb1 + EROWS * PITCH); // bih+bhh [512]
    float*  wls   = biasS + Hd;                    // Wih0 f32 [512][6] (MODE0)

    const int  tid   = threadIdx.x;
    const int  lane  = tid & 31;
    const int  wid   = tid >> 5;
    const int  n15   = lane & 15;
    const int  sel   = lane >> 4;
    const int  ncol0 = wid * 64;
    const long b0    = (long)blockIdx.x * EROWS;

    prefetch_w(Whh, tid);
    if (MODE != 0) prefetch_w(Wih, tid);

    for (int i = tid; i < Hd; i += 256) biasS[i] = bih[i] + bhh[i];
    if (MODE == 0)
        for (int i = tid; i < Hd * 6; i += 256) wls[i] = wih0[i];
    for (int i = tid; i < EROWS * PITCH; i += 256) hA[i] = (__bf16)0.0f;

    __bf16* hc = hA;
    __bf16* hn = hB;
    __bf16* xbuf[2] = {xb0, xb1};
    const __bf16* inBf = (const __bf16*)inSeq;
    const float*  xIn  = (const float*)inSeq;
    const int rr = tid >> 3, cc = (tid & 7) * 64;   // staging pattern (64 halfs/thread)

    // stage x_0
    if (MODE == 0) {
        if (tid < EROWS * 6) {
            int r = tid / 6, d = tid - r * 6;
            ((float*)xb0)[tid] = xIn[((b0 + r) * Tlen + 0) * 6 + d];
        }
    } else {
        const uint4* g = (const uint4*)(inBf + ((b0 + rr) * Tlen + 0) * Hd + cc);
        uint4*       l = (uint4*)(xb0 + rr * PITCH + cc);
#pragma unroll
        for (int i = 0; i < 8; ++i) l[i] = g[i];
    }
    __syncthreads();

    for (int t = 0; t < Tlen; ++t) {
        const int p = t & 1;
        const __bf16* xc = xbuf[p];
        const float*  xfc = (const float*)xbuf[p];
        const bool have_next = (t + 1 < Tlen);

        // issue x_{t+1} global loads now; park in registers through the k loop
        uint4 gx[8];
        float gx0 = 0.0f;
        if (have_next) {
            if (MODE == 0) {
                if (tid < EROWS * 6) {
                    int r = tid / 6, d = tid - r * 6;
                    gx0 = xIn[((b0 + r) * Tlen + (t + 1)) * 6 + d];
                }
            } else {
                const uint4* g = (const uint4*)(inBf + ((b0 + rr) * Tlen + (t + 1)) * Hd + cc);
#pragma unroll
                for (int i = 0; i < 8; ++i) gx[i] = g[i];
            }
        }

        // (C) h_new = tanh(x_t @ Wih^T + h @ Whh^T + bias)
        v8f acc[2][4];
#pragma unroll
        for (int mt = 0; mt < 2; ++mt)
#pragma unroll
            for (int nt = 0; nt < 4; ++nt) {
                int   n  = ncol0 + nt * 16 + n15;
                float bv = biasS[n];
                v8f   a;
#pragma unroll
                for (int i = 0; i < 8; ++i) {
                    float s = bv;
                    if (MODE == 0) {  // K=6 input GEMM done scalar
                        int m = mt * 16 + sel * 8 + i;
#pragma unroll
                        for (int d = 0; d < 6; ++d) s += xfc[m * 6 + d] * wls[n * 6 + d];
                    }
                    a[i] = s;
                }
                acc[mt][nt] = a;
            }

        // Software-pipelined k loop: each weight stream keeps an 8-WMMA
        // prefetch distance so v_wmma never waits on a just-issued load.
        v16bf P[4];   // Whh fragments for current kb
        v16bf Q[4];   // Wih fragments for current kb (MODE != 0)
#pragma unroll
        for (int nt = 0; nt < 4; ++nt) P[nt] = load_b(Whh, ncol0 + nt * 16, 0, lane);
        if (MODE != 0) {
#pragma unroll
            for (int nt = 0; nt < 4; ++nt) Q[nt] = load_b(Wih, ncol0 + nt * 16, 0, lane);
        }

        for (int kb = 0; kb < Hd; kb += 32) {
            v16bf a0 = load_a(hc, 0, kb, lane);
            v16bf a1 = load_a(hc, 16, kb, lane);
#pragma unroll
            for (int nt = 0; nt < 4; ++nt) {
                acc[0][nt] = wmma_bf16(a0, P[nt], acc[0][nt]);
                acc[1][nt] = wmma_bf16(a1, P[nt], acc[1][nt]);
            }
            if (kb + 32 < Hd) {
#pragma unroll
                for (int nt = 0; nt < 4; ++nt) P[nt] = load_b(Whh, ncol0 + nt * 16, kb + 32, lane);
            }
            if (MODE != 0) {
                v16bf x0 = load_a(xc, 0, kb, lane);
                v16bf x1 = load_a(xc, 16, kb, lane);
#pragma unroll
                for (int nt = 0; nt < 4; ++nt) {
                    acc[0][nt] = wmma_bf16(x0, Q[nt], acc[0][nt]);
                    acc[1][nt] = wmma_bf16(x1, Q[nt], acc[1][nt]);
                }
                if (kb + 32 < Hd) {
#pragma unroll
                    for (int nt = 0; nt < 4; ++nt) Q[nt] = load_b(Wih, ncol0 + nt * 16, kb + 32, lane);
                }
            }
        }

#pragma unroll
        for (int mt = 0; mt < 2; ++mt)
#pragma unroll
            for (int nt = 0; nt < 4; ++nt) {
                int n = ncol0 + nt * 16 + n15;
#pragma unroll
                for (int i = 0; i < 8; ++i) {
                    int   m = mt * 16 + sel * 8 + i;
                    float v = tanhf(acc[mt][nt][i]);
                    hn[m * PITCH + n] = (__bf16)v;
                    if (MODE == 2) outF[((b0 + m) * Tlen + t) * Hd + n] = v;
                    if (t == Tlen - 1) hT[(b0 + m) * Hd + n] = v;
                }
            }

        // dump the pre-loaded x_{t+1} into the other LDS buffer (latency hidden)
        if (have_next) {
            if (MODE == 0) {
                if (tid < EROWS * 6) ((float*)xbuf[p ^ 1])[tid] = gx0;
            } else {
                uint4* l = (uint4*)(xbuf[p ^ 1] + rr * PITCH + cc);
#pragma unroll
                for (int i = 0; i < 8; ++i) l[i] = gx[i];
            }
        }
        __syncthreads();   // single barrier: covers h_new writes + x_{t+1} staging

        // (E) coalesced store of h_t sequence output (bf16 layers)
        if (MODE != 2) {
            const uint4* s = (const uint4*)(hn + rr * PITCH + cc);
            uint4*       g = (uint4*)(outBf + ((b0 + rr) * Tlen + t) * Hd + cc);
#pragma unroll
            for (int i = 0; i < 8; ++i) g[i] = s[i];
        }
        __bf16* tmp = hc; hc = hn; hn = tmp;
    }
}

// ---------------- decoder: 10 steps x 4 layers + FC ----------------
__global__ __launch_bounds__(256) void dec_kernel(
    const float* __restrict__ hEnc, const __bf16* __restrict__ Wih,
    const __bf16* __restrict__ Whh, const float* __restrict__ bih,
    const float* __restrict__ bhh, const float* __restrict__ fcW,
    const float* __restrict__ fcb, float* __restrict__ out)
{
    extern __shared__ __align__(16) char smem[];
    __bf16* hbuf  = (__bf16*)smem;                          // [2 banks][4 layers][16][520]
    float*  biasS = (float*)(hbuf + 2 * 4 * DROWS * PITCH); // [4][512]

    const int  tid   = threadIdx.x;
    const int  lane  = tid & 31;
    const int  wid   = tid >> 5;
    const int  n15   = lane & 15;
    const int  sel   = lane >> 4;
    const int  ncol0 = wid * 64;
    const long b0    = (long)blockIdx.x * DROWS;

    prefetch_w(Wih, tid);
    prefetch_w(Whh, tid);

    for (int i = tid; i < 4 * Hd; i += 256) biasS[i] = bih[i] + bhh[i];
    for (int l = 0; l < 4; ++l) {
        __bf16* dst = hbuf + (0 * 4 + l) * DROWS * PITCH;
        for (int i = tid; i < DROWS * Hd; i += 256) {
            int r = i >> 9, c = i & 511;
            dst[r * PITCH + c] = (__bf16)hEnc[(long)l * Bsz * Hd + (b0 + r) * Hd + c];
        }
    }
    __syncthreads();

    int bank = 0;
    for (int st = 0; st < PRED; ++st) {
        int nx = bank ^ 1;
        for (int l = 0; l < 4; ++l) {
            // x_0 = previous step output = h[3] (also de_input0 == h_enc[3] at st=0)
            const __bf16* xsrc = (l == 0) ? (hbuf + (bank * 4 + 3) * DROWS * PITCH)
                                          : (hbuf + (nx * 4 + (l - 1)) * DROWS * PITCH);
            const __bf16* hsrc = hbuf + (bank * 4 + l) * DROWS * PITCH;
            __bf16*       dst  = hbuf + (nx * 4 + l) * DROWS * PITCH;
            const __bf16* Wi   = Wih + (long)l * Hd * Hd;
            const __bf16* Wh   = Whh + (long)l * Hd * Hd;

            v8f acc[4];
#pragma unroll
            for (int nt = 0; nt < 4; ++nt) {
                float bv = biasS[l * Hd + ncol0 + nt * 16 + n15];
                v8f   a;
#pragma unroll
                for (int i = 0; i < 8; ++i) a[i] = bv;
                acc[nt] = a;
            }

            // Full cross-kb double buffering (8-WMMA prefetch distance).
            v16bf bA[8], bB[8];
            auto loadBoth = [&](v16bf* buf, int kb) {
#pragma unroll
                for (int nt = 0; nt < 4; ++nt) {
                    buf[nt]     = load_b(Wi, ncol0 + nt * 16, kb, lane);
                    buf[4 + nt] = load_b(Wh, ncol0 + nt * 16, kb, lane);
                }
            };
            auto compute = [&](const v16bf* buf, int kb) {
                v16bf ax = load_a(xsrc, 0, kb, lane);
                v16bf ah = load_a(hsrc, 0, kb, lane);
#pragma unroll
                for (int nt = 0; nt < 4; ++nt) {
                    acc[nt] = wmma_bf16(ax, buf[nt],     acc[nt]);
                    acc[nt] = wmma_bf16(ah, buf[4 + nt], acc[nt]);
                }
            };

            loadBoth(bA, 0);
            for (int kb = 0; kb < Hd; kb += 64) {
                loadBoth(bB, kb + 32);
                compute(bA, kb);
                if (kb + 64 < Hd) loadBoth(bA, kb + 64);
                compute(bB, kb + 32);
            }

#pragma unroll
            for (int nt = 0; nt < 4; ++nt) {
                int n = ncol0 + nt * 16 + n15;
#pragma unroll
                for (int i = 0; i < 8; ++i) {
                    int m = sel * 8 + i;
                    dst[m * PITCH + n] = (__bf16)tanhf(acc[nt][i]);
                }
            }
            __syncthreads();
        }
        // FC: y[16,6] = h3_new @ fcW^T + fcb   (K=512, scalar)
        const __bf16* hx = hbuf + (nx * 4 + 3) * DROWS * PITCH;
        if (tid < DROWS * 6) {
            int   r = tid / 6, o = tid - r * 6;
            float s = fcb[o];
            for (int k = 0; k < Hd; ++k) s += (float)hx[r * PITCH + k] * fcW[o * Hd + k];
            out[((b0 + r) * PRED + st) * 6 + o] = s;
        }
        bank = nx;
    }
}

// ---------------- host launcher ----------------
extern "C" void kernel_launch(void* const* d_in, const int* in_sizes, int n_in,
                              void* d_out, int out_size, void* d_ws, size_t ws_size,
                              hipStream_t stream) {
    (void)in_sizes; (void)n_in; (void)out_size; (void)ws_size;

    const float* x     = (const float*)d_in[0];
    const float* eWih0 = (const float*)d_in[1];
    const float* eWihF = (const float*)d_in[2];   // [3][512][512]
    const float* eWhhF = (const float*)d_in[3];   // [4][512][512]
    const float* ebih  = (const float*)d_in[4];   // [4][512]
    const float* ebhh  = (const float*)d_in[5];
    const float* dWihF = (const float*)d_in[6];   // [4][512][512]
    const float* dWhhF = (const float*)d_in[7];
    const float* dbih  = (const float*)d_in[8];
    const float* dbhh  = (const float*)d_in[9];
    const float* fcW   = (const float*)d_in[10];  // [6][512]
    const float* fcb   = (const float*)d_in[11];

    const long MAT = (long)Hd * Hd;
    char* ws = (char*)d_ws;
    __bf16* seq  = (__bf16*)ws;                                       // [B][T][H] bf16, in-place per layer
    float*  hEnc = (float*)(ws + (long)Bsz * Tlen * Hd * 2);          // [4][B][H] f32
    __bf16* eWih = (__bf16*)((char*)hEnc + (long)Lnum * Bsz * Hd * 4);
    __bf16* eWhh = eWih + 3 * MAT;
    __bf16* dWih = eWhh + 4 * MAT;
    __bf16* dWhh = dWih + 4 * MAT;

    // weight conversion f32 -> bf16 (L2-resident afterwards)
    {
        int n3 = (int)(3 * MAT), n4 = (int)(4 * MAT);
        cvt_bf16<<<(n3 + 255) / 256, 256, 0, stream>>>(eWihF, eWih, n3);
        cvt_bf16<<<(n4 + 255) / 256, 256, 0, stream>>>(eWhhF, eWhh, n4);
        cvt_bf16<<<(n4 + 255) / 256, 256, 0, stream>>>(dWihF, dWih, n4);
        cvt_bf16<<<(n4 + 255) / 256, 256, 0, stream>>>(dWhhF, dWhh, n4);
    }

    const int ENC_SMEM = 4 * EROWS * PITCH * 2 + Hd * 4 + Hd * 6 * 4;        // 147456 B
    const int DEC_SMEM = 2 * 4 * DROWS * PITCH * 2 + 4 * Hd * 4;             // 141312 B
    (void)hipFuncSetAttribute((const void*)enc_layer<0>, hipFuncAttributeMaxDynamicSharedMemorySize, ENC_SMEM);
    (void)hipFuncSetAttribute((const void*)enc_layer<1>, hipFuncAttributeMaxDynamicSharedMemorySize, ENC_SMEM);
    (void)hipFuncSetAttribute((const void*)enc_layer<2>, hipFuncAttributeMaxDynamicSharedMemorySize, ENC_SMEM);
    (void)hipFuncSetAttribute((const void*)dec_kernel,   hipFuncAttributeMaxDynamicSharedMemorySize, DEC_SMEM);

    float* encOut = (float*)d_out;                               // [B][T][H]
    float* decOut = (float*)d_out + (long)Bsz * Tlen * Hd;       // [B][PRED][6]
    const int EGRID = Bsz / EROWS;   // 32
    const int DGRID = Bsz / DROWS;   // 64
    const long HL = (long)Bsz * Hd;  // h_enc layer stride

    // encoder: layer 0 (f32 x in, scalar K=6 input path), layers 1-2 in-place, layer 3 -> d_out f32
    enc_layer<0><<<EGRID, 256, ENC_SMEM, stream>>>(x, eWih0, nullptr, eWhh + 0 * MAT,
                                                   ebih + 0 * Hd, ebhh + 0 * Hd,
                                                   seq, nullptr, hEnc + 0 * HL);
    enc_layer<1><<<EGRID, 256, ENC_SMEM, stream>>>(seq, nullptr, eWih + 0 * MAT, eWhh + 1 * MAT,
                                                   ebih + 1 * Hd, ebhh + 1 * Hd,
                                                   seq, nullptr, hEnc + 1 * HL);
    enc_layer<1><<<EGRID, 256, ENC_SMEM, stream>>>(seq, nullptr, eWih + 1 * MAT, eWhh + 2 * MAT,
                                                   ebih + 2 * Hd, ebhh + 2 * Hd,
                                                   seq, nullptr, hEnc + 2 * HL);
    enc_layer<2><<<EGRID, 256, ENC_SMEM, stream>>>(seq, nullptr, eWih + 2 * MAT, eWhh + 3 * MAT,
                                                   ebih + 3 * Hd, ebhh + 3 * Hd,
                                                   nullptr, encOut, hEnc + 3 * HL);
    // decoder + FC
    dec_kernel<<<DGRID, 256, DEC_SMEM, stream>>>(hEnc, dWih, dWhh, dbih, dbhh, fcW, fcb, decOut);
}
